// DiagnosticSamplesCombine_919123002012
// MI455X (gfx1250) — matrix operand order repaired and checked by reading, not compiled, not run on hardware
//
#include <hip/hip_runtime.h>
#include <hip/hip_bf16.h>

// ---------------- problem constants ----------------
#define TT   8192
#define EE   2048
#define HH   1024
#define TH3  3072          // 3*H
#define SCAN_WGS 12        // 12 WGs * 256 threads = 3072 columns

typedef __attribute__((ext_vector_type(16))) __bf16 v16bf;
typedef __attribute__((ext_vector_type(8)))  float  v8f;

union Frag16 { unsigned int u[8]; v16bf v; };

__device__ __forceinline__ unsigned short f32_to_bf16(float f) {
    unsigned int b = __float_as_uint(f);
    return (unsigned short)((b + 0x7fffu + ((b >> 16) & 1u)) >> 16);
}

// ---------------- init: zero sync counters + out_d accumulator ----------------
__global__ void init_kernel(int* __restrict__ sync, float* __restrict__ out_d) {
    int i = blockIdx.x * 256 + threadIdx.x;
    if (i < TT)   sync[i]  = 0;
    if (i < EE)   out_d[i] = 0.0f;
}

// ---------------- f32 -> bf16 (RNE), row-major copy (for emb_seq) ----------------
__global__ void cast_kernel(const float* __restrict__ in, unsigned short* __restrict__ out, int n) {
    int i = blockIdx.x * blockDim.x + threadIdx.x;
    int stride = gridDim.x * blockDim.x;
    for (; i < n; i += stride) out[i] = f32_to_bf16(in[i]);
}

// ---------------- f32 [2048][3072] -> bf16 transposed [3072][2048] (for w_i_e) ----------------
__global__ __launch_bounds__(256) void castT_kernel(const float* __restrict__ in,
                                                    unsigned short* __restrict__ out) {
    __shared__ unsigned short tile[64][65];          // padded to dodge bank conflicts
    const int bk = blockIdx.x * 64;                  // k base (2048/64 = 32)
    const int bn = blockIdx.y * 64;                  // n base (3072/64 = 48)
    const int c  = threadIdx.x & 63;
    const int r0 = threadIdx.x >> 6;                 // 0..3
    #pragma unroll
    for (int r = 0; r < 16; ++r) {
        int k = r0 + 4 * r;
        tile[k][c] = f32_to_bf16(in[(size_t)(bk + k) * TH3 + bn + c]);
    }
    __syncthreads();
    #pragma unroll
    for (int r = 0; r < 16; ++r) {
        int n = r0 + 4 * r;
        out[(size_t)(bn + n) * EE + bk + c] = tile[c][n];
    }
}

// ---------------- out_d = diag_seq.sum(axis=0)  (softmax over size-1 axis == 1) ----------------
__global__ void colsum_kernel(const float* __restrict__ diag, float* __restrict__ out_d) {
    int e  = blockIdx.x * 256 + threadIdx.x;   // gridDim.x = 8  -> 2048 columns
    int tc = blockIdx.y;                       // gridDim.y = 32 -> 256 rows per chunk
    float s = 0.0f;
    int t0 = tc * 256;
    #pragma unroll 4
    for (int t = t0; t < t0 + 256; ++t)
        s += diag[(size_t)t * EE + e];
    atomicAdd(&out_d[e], s);
}

// ---------------- WMMA GEMM: gx[8192x3072] = emb_bf16[8192x2048] @ wiT_bf16[3072x2048]^T --------
// Block tile 128x64, K-step 32, double-buffered LDS.  8 waves in a 4x2 grid; each wave owns a
// 32x32 tile = 2 A-frags x 2 B-frags -> 4 WMMAs per 8 ds_load_b128.  The K-loop is pinned to
// unroll 1: the manual double-buffer already overlaps global/LDS with WMMA, and letting the
// compiler unroll x2 made it rotate accumulator register sets (8-VGPR v_mov chains + 4 forced
// hazard v_nops per WMMA at the unroll seam).  unroll(disable) keeps D==C in-place accumulation.
__global__ __launch_bounds__(256) void gemm_kernel(const unsigned short* __restrict__ A,
                                                   const unsigned short* __restrict__ Bt,
                                                   float* __restrict__ C) {
    __shared__ unsigned int ldsA[2][128 * 16];   // [buf][row(128) x 32 bf16]
    __shared__ unsigned int ldsB[2][64 * 16];    // [buf][n(64)    x 32 bf16]

    const int tid  = threadIdx.x;
    const int lane = tid & 31;
    const int wave = tid >> 5;
    const int m0   = blockIdx.x * 128;
    const int n0   = blockIdx.y * 64;
    const int wr   = (wave & 3) * 32;            // wave row offset   (4 x 32 = 128)
    const int wc   = (wave >> 2) * 32;           // wave col offset   (2 x 32 = 64)

    // cooperative staging: uint4 = 8 bf16; A tile = 512 uint4 (2/thread), B tile = 256 (1/thread)
    const int arow = tid >> 2, ag = tid & 3;
    const int brow = tid >> 2, bg = tid & 3;
    const uint4* Ag4 = (const uint4*)A;          // row pitch = EE/8 = 256 uint4
    const uint4* Bg4 = (const uint4*)Bt;

    v8f c00 = {}, c01 = {}, c10 = {}, c11 = {};

    uint4 ra0 = Ag4[(size_t)(m0 + arow) * 256 + ag];
    uint4 ra1 = Ag4[(size_t)(m0 + arow + 64) * 256 + ag];
    uint4 rbv = Bg4[(size_t)(n0 + brow) * 256 + bg];
    *(uint4*)&ldsA[0][arow * 16 + ag * 4]        = ra0;
    *(uint4*)&ldsA[0][(arow + 64) * 16 + ag * 4] = ra1;
    *(uint4*)&ldsB[0][brow * 16 + bg * 4]        = rbv;
    __syncthreads();

    const int NITER  = EE / 32;                  // 64 K-steps
    const int a0base = (wr + (lane & 15)) * 16 + (lane >> 4) * 8;
    const int a1base = a0base + 16 * 16;
    const int b0base = (wc + (lane & 15)) * 16 + (lane >> 4) * 8;
    const int b1base = b0base + 16 * 16;

    #pragma clang loop unroll(disable)
    for (int i = 0; i < NITER; ++i) {
        const unsigned int* la = ldsA[i & 1];
        const unsigned int* lb = ldsB[i & 1];

        Frag16 a0, a1, b0, b1;
        #pragma unroll
        for (int v = 0; v < 8; ++v) {
            int o = v;
            a0.u[v] = la[a0base + o];
            a1.u[v] = la[a1base + o];
        }
        #pragma unroll
        for (int v = 0; v < 8; ++v) { b0.u[v] = lb[b0base + v]; b1.u[v] = lb[b1base + v]; }

        // issue next tile's global loads BEFORE the WMMAs so they overlap matrix math
        if (i + 1 < NITER) {
            int k4 = (i + 1) * 4;
            ra0 = Ag4[(size_t)(m0 + arow) * 256 + k4 + ag];
            ra1 = Ag4[(size_t)(m0 + arow + 64) * 256 + k4 + ag];
            rbv = Bg4[(size_t)(n0 + brow) * 256 + k4 + bg];
        }

        c00 = __builtin_amdgcn_wmma_f32_16x16x32_bf16(false, a0.v, false, b0.v, (short)0, c00, false, false);
        c01 = __builtin_amdgcn_wmma_f32_16x16x32_bf16(false, a0.v, false, b1.v, (short)0, c01, false, false);
        c10 = __builtin_amdgcn_wmma_f32_16x16x32_bf16(false, a1.v, false, b0.v, (short)0, c10, false, false);
        c11 = __builtin_amdgcn_wmma_f32_16x16x32_bf16(false, a1.v, false, b1.v, (short)0, c11, false, false);

        if (i + 1 < NITER) {
            int nb = (i + 1) & 1;
            *(uint4*)&ldsA[nb][arow * 16 + ag * 4]        = ra0;
            *(uint4*)&ldsA[nb][(arow + 64) * 16 + ag * 4] = ra1;
            *(uint4*)&ldsB[nb][brow * 16 + bg * 4]        = rbv;
        }
        __syncthreads();
    }

    // store: accumulator VGPR g holds row +g (lanes<16) / +8+g (lanes>=16), col = lane&15
    const int cb  = n0 + wc + (lane & 15);
    const int rb0 = m0 + wr + (lane >> 4) * 8;
    #pragma unroll
    for (int g = 0; g < 8; ++g) {
        size_t r0 = (size_t)(rb0 + g) * TH3;
        size_t r1 = (size_t)(rb0 + 16 + g) * TH3;
        C[r0 + cb]      = c00[g];
        C[r0 + cb + 16] = c01[g];
        C[r1 + cb]      = c10[g];
        C[r1 + cb + 16] = c11[g];
    }
}

// ---------------- serial reverse GRU scan, matvec split over 12 WGs ----------------
__global__ __launch_bounds__(256) void scan_kernel(const float* __restrict__ gx,
                                                   const float* __restrict__ w_h,
                                                   const float* __restrict__ b,
                                                   const float* __restrict__ emb,
                                                   const float* __restrict__ w_att,
                                                   const float* __restrict__ b_att,
                                                   float* __restrict__ out_e,
                                                   float* dots, int* sync) {
    __shared__ float h_lds[HH];
    __shared__ float red[256];
    const int tid = threadIdx.x;
    const int wg  = blockIdx.x;
    const int col = wg * 256 + tid;            // owned output column in [0, 3072)

    for (int i = tid; i < HH; i += 256) h_lds[i] = 0.0f;
    float acc[8];
    #pragma unroll
    for (int q = 0; q < 8; ++q) acc[q] = 0.0f;
    __syncthreads();

    for (int t = TT - 1; t >= 0; --t) {
        const float* gxr = gx + (size_t)t * TH3;
        // warm L2/L0 for the phase-B stream (gx comes from HBM exactly once) while
        // the dot chain below runs against the L2-resident w_h.
        __builtin_prefetch(gxr + col, 0, 3);
        if (wg == 0) __builtin_prefetch(emb + (size_t)t * EE + tid * 8, 0, 3);

        // ---- phase A: d[col] = h . w_h[:, col]  (coalesced: column-contiguous rows) ----
        float d = 0.0f;
        const float* wcol = w_h + col;
        #pragma unroll 4
        for (int k = 0; k < HH; ++k) d = fmaf(h_lds[k], wcol[(size_t)k * TH3], d);
        float* db = dots + (t & 1) * TH3;      // double-buffered -> one sync/step
        db[col] = d;

        // ---- grid-wide sync (release/acquire) ----
        __threadfence();
        __syncthreads();
        if (tid == 0) {
            atomicAdd(&sync[t], 1);
            while (__hip_atomic_load(&sync[t], __ATOMIC_RELAXED, __HIP_MEMORY_SCOPE_AGENT) < SCAN_WGS) {}
        }
        __syncthreads();
        __threadfence();

        // ---- phase B: gate update (redundant in every WG; keeps h local in LDS) ----
        float attp = 0.0f;
        #pragma unroll
        for (int q = 0; q < 4; ++q) {
            int i = tid + q * 256;
            float zp = gxr[i]          + db[i]          + b[i];
            float rp = gxr[HH + i]     + db[HH + i]     + b[HH + i];
            float z  = 1.0f / (1.0f + __expf(-zp));
            float r  = 1.0f / (1.0f + __expf(-rp));
            float ap = gxr[2*HH + i] + r * db[2*HH + i] + b[2*HH + i];
            float a  = tanhf(ap);
            float hn = (1.0f - z) * h_lds[i] + z * a;
            h_lds[i] = hn;
            attp = fmaf(hn, w_att[i], attp);
        }
        __syncthreads();

        if (wg == 0) {
            red[tid] = attp;
            __syncthreads();
            for (int s = 128; s > 0; s >>= 1) {
                if (tid < s) red[tid] += red[tid + s];
                __syncthreads();
            }
            float att = tanhf(red[0] + b_att[0]);
            const float* er = emb + (size_t)t * EE;
            #pragma unroll
            for (int q = 0; q < 8; ++q) acc[q] = fmaf(att, er[tid + q * 256], acc[q]);
            __syncthreads();
        }
        __syncthreads();
    }

    if (wg == 0) {
        #pragma unroll
        for (int q = 0; q < 8; ++q) out_e[tid + q * 256] = acc[q];
    }
}

// ---------------- launcher ----------------
extern "C" void kernel_launch(void* const* d_in, const int* in_sizes, int n_in,
                              void* d_out, int out_size, void* d_ws, size_t ws_size,
                              hipStream_t stream) {
    const float* emb_seq = (const float*)d_in[0];   // [8192,2048]
    const float* diag    = (const float*)d_in[1];   // [8192,2048]
    const float* w_i_e   = (const float*)d_in[2];   // [2048,3072]
    const float* w_h_e   = (const float*)d_in[3];   // [1024,3072]
    const float* b_e     = (const float*)d_in[4];   // [3072]
    // d_in[5..7] (GRU-d) are dead: softmax over a size-1 axis is identically 1.
    const float* w_att_e = (const float*)d_in[8];   // [1024]
    const float* b_att_e = (const float*)d_in[9];   // [1]

    float* out_e = (float*)d_out;                   // [2048]
    float* out_d = (float*)d_out + EE;              // [2048]

    char* ws = (char*)d_ws;
    unsigned short* emb_bf = (unsigned short*)ws;                               // 33,554,432 B
    unsigned short* wiT_bf = (unsigned short*)(ws + 33554432);                  // 12,582,912 B (transposed)
    float*          gx     = (float*)(ws + 46137344);                           // 100,663,296 B
    float*          dots   = (float*)(ws + 146800640);                          // 24,576 B
    int*            syncb  = (int*)  (ws + 146825216);                          // 32,768 B

    init_kernel<<<32, 256, 0, stream>>>(syncb, out_d);
    cast_kernel<<<1024, 256, 0, stream>>>(emb_seq, emb_bf, TT * EE);
    castT_kernel<<<dim3(EE / 64, TH3 / 64), 256, 0, stream>>>(w_i_e, wiT_bf);
    colsum_kernel<<<dim3(8, 32), 256, 0, stream>>>(diag, out_d);

    gemm_kernel<<<dim3(TT / 128, TH3 / 64), 256, 0, stream>>>(emb_bf, wiT_bf, gx);

    scan_kernel<<<SCAN_WGS, 256, 0, stream>>>(gx, w_h_e, b_e, emb_seq,
                                              w_att_e, b_att_e, out_e, dots, syncb);
}
